// MultiHeadAttention_58342835749671
// MI455X (gfx1250) — matrix operand order repaired and checked by reading, not compiled, not run on hardware
//
#include <hip/hip_runtime.h>
#include <hip/hip_bf16.h>
#include <cmath>

#define BB 2
#define SS 2048
#define DMODEL 1024
#define NH 16
#define HDIM 64

typedef __attribute__((ext_vector_type(16))) __bf16 v16bf;
typedef __attribute__((ext_vector_type(8)))  float  v8f;

// Store modes for the GEMM epilogue
#define MODE_F32 0   // fp32, row-major [M][N]
#define MODE_B16 1   // bf16, row-major [M][N]
#define MODE_VT  2   // bf16, head-transposed [b][h][hd][s]

// ---- WMMA fragment index helpers (contiguous k order chosen by this file, not the ISA table order; 16-bit, wave32) ----
// A matrix 16x32 (MxK): lane L holds row m = L%16, element i -> k = i + 16*(L/16)
__device__ __forceinline__ int a_koff(int lane, int i) {
  return i + ((lane & 16) ? 16 : 0);
}
// B matrix 32x16 (KxN): lane L holds col n = L%16, element i -> k = i + 16*(L/16)
__device__ __forceinline__ int b_koff(int lane, int i) {
  return i + ((lane & 16) ? 16 : 0);
}

// ---------------------------------------------------------------------------
// One-shot fp32 -> bf16 convert (row-major, 8 elements/thread).
// ---------------------------------------------------------------------------
__global__ __launch_bounds__(256)
void cvt_bf16(const float* __restrict__ in, __bf16* __restrict__ out, int n) {
  const int base = (blockIdx.x * 256 + threadIdx.x) * 8;
  if (base >= n) return;
#pragma unroll
  for (int i = 0; i < 8; ++i) out[base + i] = (__bf16)in[base + i];
}

// ---------------------------------------------------------------------------
// One-shot fp32 -> bf16 convert + transpose: in[K][N] -> out[N][K].
// ---------------------------------------------------------------------------
__global__ __launch_bounds__(256)
void cvt_bf16_t(const float* __restrict__ in, __bf16* __restrict__ out,
                int K, int N) {
  const int base = (blockIdx.x * 256 + threadIdx.x) * 4;
  if (base >= N * K) return;
  const int nn = base / K;
  const int k0 = base % K;
#pragma unroll
  for (int i = 0; i < 4; ++i)
    out[(size_t)nn * K + k0 + i] = (__bf16)in[(size_t)(k0 + i) * N + nn];
}

// ---------------------------------------------------------------------------
// Register-blocked bf16 GEMM: out[M,N] = A[M,K] * Bt[N,K]^T, fp32 accumulate.
// Both operands bf16, all fragment loads contiguous (2x b128 each).
// Each wave computes a 32x64 strip (2 M-frags x 4 N-frags -> 8 WMMAs per
// 32-wide K step). 4 waves/block. Store layout selected at compile time.
// ---------------------------------------------------------------------------
template <int MODE>
__global__ __launch_bounds__(128)
void gemm_bf16wmma(const __bf16* __restrict__ A, const __bf16* __restrict__ Bt,
                   float* __restrict__ outF, __bf16* __restrict__ outB,
                   int M, int N, int K, float scale) {
  const int lane  = threadIdx.x & 31;
  const int wv    = threadIdx.x >> 5;
  const int strip = blockIdx.x * 4 + wv;
  const int stripsN = N >> 6;                 // strips of 64 cols
  const int sm = strip / stripsN;
  const int sn = strip % stripsN;
  if (sm * 32 >= M) return;

  const int l15 = lane & 15;
  const int m0  = sm * 32;
  const int n0  = sn * 64;
  const int bko = (lane & 16) ? 16 : 0;       // B-frag k base for this half

  v8f c[2][4];
#pragma unroll
  for (int mt = 0; mt < 2; ++mt)
#pragma unroll
    for (int nt = 0; nt < 4; ++nt)
      c[mt][nt] = (v8f){0.f,0.f,0.f,0.f,0.f,0.f,0.f,0.f};

  for (int kk = 0; kk < K; kk += 32) {
    v16bf a[2], b[4];
#pragma unroll
    for (int mt = 0; mt < 2; ++mt) {
      const __bf16* ap = A + (size_t)(m0 + mt * 16 + l15) * K + kk;
#pragma unroll
      for (int i = 0; i < 16; ++i) a[mt][i] = ap[a_koff(lane, i)];
    }
#pragma unroll
    for (int nt = 0; nt < 4; ++nt) {
      const __bf16* bp = Bt + (size_t)(n0 + nt * 16 + l15) * K + kk + bko;
#pragma unroll
      for (int i = 0; i < 16; ++i) b[nt][i] = bp[i];   // 16 contiguous bf16
    }
#pragma unroll
    for (int mt = 0; mt < 2; ++mt)
#pragma unroll
      for (int nt = 0; nt < 4; ++nt)
        c[mt][nt] = __builtin_amdgcn_wmma_f32_16x16x32_bf16(
            false, a[mt], false, b[nt], (short)0, c[mt][nt], false, false);
  }

#pragma unroll
  for (int mt = 0; mt < 2; ++mt)
#pragma unroll
    for (int nt = 0; nt < 4; ++nt) {
      const int col = n0 + nt * 16 + l15;
#pragma unroll
      for (int j = 0; j < 8; ++j) {
        const int m = m0 + mt * 16 + j + ((lane & 16) ? 8 : 0);
        const float v = c[mt][nt][j] * scale;
        if (MODE == MODE_F32) {
          outF[(size_t)m * N + col] = v;
        } else if (MODE == MODE_B16) {
          outB[(size_t)m * N + col] = (__bf16)v;
        } else {                                  // MODE_VT: [b][h][hd][s]
          const int bb = m / SS, s = m % SS;
          const int h = col / HDIM, hd = col % HDIM;
          outB[(((size_t)bb * NH + h) * HDIM + hd) * SS + s] = (__bf16)v;
        }
      }
    }
}

// ---------------------------------------------------------------------------
// Flash attention: one wave per (b, h, 16-row q-tile). Streams 32-key tiles.
// Q pre-scaled by 1/sqrt(hd). V is head-transposed [b][h][hd][s] so PV
// B-fragments are contiguous. Output bf16 [b*s, DMODEL] (feeds Wo GEMM).
// ---------------------------------------------------------------------------
__global__ __launch_bounds__(128)
void flash_attn(const __bf16* __restrict__ Q, const __bf16* __restrict__ Kb,
                const __bf16* __restrict__ Vt, const int* __restrict__ amask,
                const int* __restrict__ causal_p, __bf16* __restrict__ Ob) {
  __shared__ __bf16 p_lds[4][16][32];     // wave-private 16x32 P staging

  const int lane = threadIdx.x & 31;
  const int wv   = threadIdx.x >> 5;
  const int t    = blockIdx.x * 4 + wv;
  const int qtiles = SS / 16;
  const int bh = t / qtiles;
  const int qt = t % qtiles;
  const int b  = bh / NH;
  const int h  = bh % NH;
  const int q0 = qt * 16;
  const int causal = *causal_p;
  const int l15 = lane & 15;
  const int bko = (lane & 16) ? 16 : 0;
  const size_t bh_off = (size_t)b * SS * DMODEL + h * HDIM;
  const size_t v_off  = ((size_t)b * NH + h) * HDIM * SS;   // Vt head base

  // Q A-fragments: two 32-wide hd chunks
  v16bf aq[2];
  {
    const __bf16* qrow = Q + bh_off + (size_t)(q0 + l15) * DMODEL;
#pragma unroll
    for (int ch = 0; ch < 2; ++ch)
#pragma unroll
      for (int i = 0; i < 16; ++i)
        aq[ch][i] = qrow[ch * 32 + a_koff(lane, i)];
  }

  v8f o[4];
#pragma unroll
  for (int f = 0; f < 4; ++f) o[f] = (v8f){0.f,0.f,0.f,0.f,0.f,0.f,0.f,0.f};
  float mi[8], li[8];
#pragma unroll
  for (int j = 0; j < 8; ++j) { mi[j] = -INFINITY; li[j] = 0.f; }

  const int ntiles = causal ? ((q0 + 16 + 31) >> 5) : (SS >> 5);

  for (int kt = 0; kt < ntiles; ++kt) {
    const int k_base = kt * 32;

    // ---- scores S = Q * K^T (16q x 32k), two 16-key B-fragments ----
    v8f s[2];
#pragma unroll
    for (int f = 0; f < 2; ++f) {
      s[f] = (v8f){0.f,0.f,0.f,0.f,0.f,0.f,0.f,0.f};
      const int key = k_base + f * 16 + l15;
      const __bf16* krow = Kb + bh_off + (size_t)key * DMODEL + bko;
      v16bf bk0, bk1;
#pragma unroll
      for (int i = 0; i < 16; ++i) { bk0[i] = krow[i]; bk1[i] = krow[32 + i]; }
      s[f] = __builtin_amdgcn_wmma_f32_16x16x32_bf16(false, aq[0], false, bk0,
                                                     (short)0, s[f], false, false);
      s[f] = __builtin_amdgcn_wmma_f32_16x16x32_bf16(false, aq[1], false, bk1,
                                                     (short)0, s[f], false, false);
    }

    // ---- masking + online softmax (row m = q0 + j + 8*(lane/16), col = lane%16) ----
    const int n0 = k_base + l15;
    const int n1 = n0 + 16;
    const int am0 = amask[b * SS + n0];
    const int am1 = amask[b * SS + n1];
#pragma unroll
    for (int j = 0; j < 8; ++j) {
      const int m = q0 + j + ((lane & 16) ? 8 : 0);
      float x0 = s[0][j], x1 = s[1][j];
      if ((causal && n0 > m) || !am0) x0 = -INFINITY;
      if ((causal && n1 > m) || !am1) x1 = -INFINITY;
      float mx = fmaxf(x0, x1);
#pragma unroll
      for (int d = 1; d < 16; d <<= 1) mx = fmaxf(mx, __shfl_xor(mx, d, 32));
      const float mnew = fmaxf(mi[j], mx);
      float alpha, p0, p1;
      if (mnew == -INFINITY) { alpha = 1.f; p0 = 0.f; p1 = 0.f; }
      else {
        alpha = __expf(mi[j] - mnew);
        p0 = (x0 == -INFINITY) ? 0.f : __expf(x0 - mnew);
        p1 = (x1 == -INFINITY) ? 0.f : __expf(x1 - mnew);
      }
      float rs = p0 + p1;
#pragma unroll
      for (int d = 1; d < 16; d <<= 1) rs += __shfl_xor(rs, d, 32);
      li[j] = li[j] * alpha + rs;
      mi[j] = mnew;
#pragma unroll
      for (int f = 0; f < 4; ++f) o[f][j] *= alpha;
      const int mrow = j + ((lane & 16) ? 8 : 0);
      p_lds[wv][mrow][l15]      = (__bf16)p0;
      p_lds[wv][mrow][l15 + 16] = (__bf16)p1;
    }

    // wave-private LDS: ensure our stores land before cross-lane reads
    asm volatile("s_wait_dscnt 0" ::: "memory");

    // P (C-layout) -> A-fragment layout via LDS
    v16bf ap_;
    {
      const __bf16* prow = &p_lds[wv][l15][0];
#pragma unroll
      for (int i = 0; i < 16; ++i) ap_[i] = prow[a_koff(lane, i)];
    }

    // ---- O += P * V : 4 hd tiles of 16, K = 32 keys (contiguous Vt loads) ----
#pragma unroll
    for (int f = 0; f < 4; ++f) {
      const __bf16* vrow = Vt + v_off + (size_t)(f * 16 + l15) * SS + k_base + bko;
      v16bf bv;
#pragma unroll
      for (int i = 0; i < 16; ++i) bv[i] = vrow[i];    // 16 contiguous bf16
      o[f] = __builtin_amdgcn_wmma_f32_16x16x32_bf16(false, ap_, false, bv,
                                                     (short)0, o[f], false, false);
    }
  }

  // ---- normalize and write bf16 [b*s, DMODEL] ----
#pragma unroll
  for (int j = 0; j < 8; ++j) {
    const int m = q0 + j + ((lane & 16) ? 8 : 0);
    const float inv = (li[j] > 0.f) ? (1.f / li[j]) : 0.f;
    __bf16* orow = Ob + ((size_t)b * SS + m) * DMODEL + h * HDIM + l15;
    orow[0]  = (__bf16)(o[0][j] * inv);
    orow[16] = (__bf16)(o[1][j] * inv);
    orow[32] = (__bf16)(o[2][j] * inv);
    orow[48] = (__bf16)(o[3][j] * inv);
  }
}

// ---------------------------------------------------------------------------
extern "C" void kernel_launch(void* const* d_in, const int* in_sizes, int n_in,
                              void* d_out, int out_size, void* d_ws, size_t ws_size,
                              hipStream_t stream) {
  const float* x      = (const float*)d_in[0];
  const int*   amask  = (const int*)d_in[1];
  const int*   causal = (const int*)d_in[2];
  const float* Wq     = (const float*)d_in[3];
  const float* Wk     = (const float*)d_in[4];
  const float* Wv     = (const float*)d_in[5];
  const float* Wo     = (const float*)d_in[6];
  float* out = (float*)d_out;

  char* ws = (char*)d_ws;
  const size_t MB = (size_t)1 << 20;
  __bf16* xb  = (__bf16*)(ws);             // 8 MB  [M,K] row-major
  __bf16* Wqt = (__bf16*)(ws + 8  * MB);   // 2 MB  [N,K] (transposed)
  __bf16* Wkt = (__bf16*)(ws + 10 * MB);   // 2 MB
  __bf16* Wvt = (__bf16*)(ws + 12 * MB);   // 2 MB
  __bf16* Wot = (__bf16*)(ws + 14 * MB);   // 2 MB
  __bf16* Qb  = (__bf16*)(ws + 16 * MB);   // 8 MB  [b*s, d]
  __bf16* Kb  = (__bf16*)(ws + 24 * MB);   // 8 MB  [b*s, d]
  __bf16* Vtb = (__bf16*)(ws + 32 * MB);   // 8 MB  [b][h][hd][s]
  __bf16* Ab  = (__bf16*)(ws + 40 * MB);   // 8 MB  [b*s, d]

  const int M = BB * SS, N = DMODEL, K = DMODEL;
  const dim3 blk256(256), blk128(128);

  // One-shot conversions: x row-major bf16; weights transposed bf16.
  cvt_bf16<<<(M * K) / (256 * 8), blk256, 0, stream>>>(x, xb, M * K);
  cvt_bf16_t<<<(K * N) / (256 * 4), blk256, 0, stream>>>(Wq, Wqt, K, N);
  cvt_bf16_t<<<(K * N) / (256 * 4), blk256, 0, stream>>>(Wk, Wkt, K, N);
  cvt_bf16_t<<<(K * N) / (256 * 4), blk256, 0, stream>>>(Wv, Wvt, K, N);
  cvt_bf16_t<<<(K * N) / (256 * 4), blk256, 0, stream>>>(Wo, Wot, K, N);

  const int gemm_blocks = (M / 32) * (N / 64) / 4;   // 32x64 strip/wave, 4 waves/block
  const float qscale = 0.125f;                       // 1/sqrt(HDIM)
  gemm_bf16wmma<MODE_B16><<<gemm_blocks, blk128, 0, stream>>>(xb, Wqt, nullptr, Qb,  M, N, K, qscale);
  gemm_bf16wmma<MODE_B16><<<gemm_blocks, blk128, 0, stream>>>(xb, Wkt, nullptr, Kb,  M, N, K, 1.0f);
  gemm_bf16wmma<MODE_VT ><<<gemm_blocks, blk128, 0, stream>>>(xb, Wvt, nullptr, Vtb, M, N, K, 1.0f);

  const int attn_blocks = BB * NH * (SS / 16) / 4;   // 4 waves/block
  flash_attn<<<attn_blocks, blk128, 0, stream>>>(Qb, Kb, Vtb, amask, causal, Ab);

  gemm_bf16wmma<MODE_F32><<<gemm_blocks, blk128, 0, stream>>>(Ab, Wot, out, nullptr, M, N, K, 1.0f);
}
